// MaxViT_28759101014495
// MI455X (gfx1250) — matrix operand order repaired and checked by reading, not compiled, not run on hardware
//
#include <hip/hip_runtime.h>
#include <hip/hip_bf16.h>
#include <cstdint>

// ---------------- types ----------------
typedef __bf16 v16bf __attribute__((ext_vector_type(16)));
typedef __bf16 v8bf  __attribute__((ext_vector_type(8)));
typedef float  v8f   __attribute__((ext_vector_type(8)));
typedef int    v4i   __attribute__((ext_vector_type(4)));

#define TOK 65536   // B*H*W
#define CCH 256
#define CE  1024
#define NHD 8
#define HD  32

// ---------------- CDNA5 feature gates ----------------
#if defined(__HIP_DEVICE_COMPILE__) && __has_builtin(__builtin_amdgcn_global_load_async_to_lds_b128)
#define USE_ASYNC_LDS 1
#endif

__device__ __forceinline__ void cp16(void* lds, const void* g) {
#ifdef USE_ASYNC_LDS
    // async DMA global -> LDS, tracked by ASYNCcnt (gfx1250)
    __builtin_amdgcn_global_load_async_to_lds_b128(
        (__attribute__((address_space(1))) v4i*)(void*)g,
        (__attribute__((address_space(3))) v4i*)lds, 0, 0);
#else
    *(v4i*)lds = *(const v4i*)g;
#endif
}

__device__ __forceinline__ void async_wait() {
#if defined(__HIP_DEVICE_COMPILE__) && __has_builtin(__builtin_amdgcn_s_wait_asynccnt)
    __builtin_amdgcn_s_wait_asynccnt(0);
#endif
}

__device__ __forceinline__ v8f wmma_bf16(v16bf a, v16bf b, v8f c) {
#if defined(__HIP_DEVICE_COMPILE__)
    return __builtin_amdgcn_wmma_f32_16x16x32_bf16(false, a, false, b, (short)0, c,
                                                   false, false);
#else
    (void)a; (void)b; return c;
#endif
}

__device__ __forceinline__ float gelu_f(float x) {
    return 0.5f * x * (1.0f + erff(x * 0.70710678118654752f));
}

// A-fragment (16x32 bf16): lane holds row m=lane%16; elems 0..7 = K kb..kb+7,
// elems 8..15 = K kb+16..kb+23, kb = (lane>=16)*8. p points at row-base + kb.
__device__ __forceinline__ v16bf ldfragA(const __bf16* p) {
    v8bf lo = *(const v8bf*)p;
    v8bf hi = *(const v8bf*)(p + 16);
    v16bf a;
#pragma unroll
    for (int i = 0; i < 8; ++i) { a[i] = lo[i]; a[i + 8] = hi[i]; }
    return a;
}

// B-fragment (32x16 bf16): lane holds col n=lane%16; elems 0..7 = K kb..kb+7,
// elems 8..15 = K kb+16..kb+23, kb=(lane>=16)*8.  B stored transposed ([N][K]).
__device__ __forceinline__ v16bf ldfragB(const __bf16* p) {
    v8bf lo = *(const v8bf*)p;
    v8bf hi = *(const v8bf*)(p + 16);
    v16bf b;
#pragma unroll
    for (int i = 0; i < 8; ++i) { b[i] = lo[i]; b[i + 8] = hi[i]; }
    return b;
}

// ================= weight prep =================
__global__ void tcast_kernel(const float* __restrict__ in, __bf16* __restrict__ out,
                             int K, int N) {
    long idx = (long)blockIdx.x * 256 + threadIdx.x;
    if (idx >= (long)K * N) return;
    int n = (int)(idx % N); long k = idx / N;
    out[(long)n * K + k] = (__bf16)in[idx];
}

__global__ void bnfold_kernel(const float* g, const float* b, const float* m,
                              const float* v, float* sc, float* sh, int N) {
    int i = blockIdx.x * 256 + threadIdx.x;
    if (i < N) { float s = g[i] * rsqrtf(v[i] + 1e-3f); sc[i] = s; sh[i] = b[i] - m[i] * s; }
}

__global__ void bn0cast_kernel(const float* __restrict__ X, const float* g,
                               const float* b, const float* m, const float* v,
                               __bf16* __restrict__ Xb, float* __restrict__ Out) {
    long idx = (long)blockIdx.x * 256 + threadIdx.x;
    int c = (int)(idx & (CCH - 1));
    float x = X[idx];
    Out[idx] = x;                                  // residual stream init
    Xb[idx] = (__bf16)((x - m[c]) * rsqrtf(v[c] + 1e-3f) * g[c] + b[c]);
}

// ================= WMMA GEMM =================
// C[M,N] = A[M,K] * B^T[N,K],   block tile 128x64, 8 waves of 32x32, K-step 32.
// EPI: 0 store bf16, 1 (acc+bias)*ps -> bf16, 2 gelu(acc+bias) -> bf16,
//      3 gelu(acc*sc+sh) -> bf16, 4 outf += acc+bias, 5 outf += acc
template <int EPI>
__global__ __launch_bounds__(256) void gemm_bf16(
    const __bf16* __restrict__ A, const __bf16* __restrict__ BT,
    const float* __restrict__ bias, const float* __restrict__ sc,
    const float* __restrict__ sh, __bf16* __restrict__ outb,
    float* __restrict__ outf, int K, int N, float ps) {
    __shared__ __bf16 As[2][128 * 32];
    __shared__ __bf16 Bs[2][64 * 32];
    const int tid = threadIdx.x;
    const long m0 = (long)blockIdx.y * 128;
    const long n0 = (long)blockIdx.x * 64;
    const int wv = tid >> 5, lane = tid & 31;
    const int wm = wv & 3, wn = wv >> 2;
    const int lrow = lane & 15, lhi = lane >> 4;

    v8f acc[2][2] = {};

    auto load_tile = [&](int buf, int k0) {
#pragma unroll
        for (int i = 0; i < 2; ++i) {                 // A: 128 rows x 32k = 512 x 16B
            int idx = tid + i * 256;
            int row = idx >> 2, seg = idx & 3;
            cp16(&As[buf][row * 32 + seg * 8],
                 &A[(m0 + row) * K + k0 + seg * 8]);
        }
        {                                             // B^T: 64 rows x 32k = 256 x 16B
            int row = tid >> 2, seg = tid & 3;
            cp16(&Bs[buf][row * 32 + seg * 8],
                 &BT[(n0 + row) * K + k0 + seg * 8]);
        }
    };

    load_tile(0, 0);
    const int nk = K >> 5;
    for (int ks = 0; ks < nk; ++ks) {
        const int buf = ks & 1;
        async_wait();
        __syncthreads();
        if (ks + 1 < nk) {
            load_tile(buf ^ 1, (ks + 1) << 5);
            if (ks + 2 < nk)   // L2 warm-up hint -> global_prefetch
                __builtin_prefetch(&A[(m0 + (tid >> 2)) * K + ((ks + 2) << 5)], 0, 1);
        }
        v16bf af[2], bf[2];
#pragma unroll
        for (int mi = 0; mi < 2; ++mi)
            af[mi] = ldfragA(&As[buf][(wm * 32 + mi * 16 + lrow) * 32 + lhi * 8]);
#pragma unroll
        for (int ni = 0; ni < 2; ++ni)
            bf[ni] = ldfragB(&Bs[buf][(wn * 32 + ni * 16 + lrow) * 32 + lhi * 8]);
#pragma unroll
        for (int mi = 0; mi < 2; ++mi)
#pragma unroll
            for (int ni = 0; ni < 2; ++ni)
                acc[mi][ni] = wmma_bf16(af[mi], bf[ni], acc[mi][ni]);
    }

    // epilogue: C layout -> lane l, vgpr r: row = base + r + 8*(l>=16), col = base + l%16
#pragma unroll
    for (int mi = 0; mi < 2; ++mi)
#pragma unroll
        for (int ni = 0; ni < 2; ++ni)
#pragma unroll
            for (int r = 0; r < 8; ++r) {
                long row = m0 + wm * 32 + mi * 16 + r + 8 * lhi;
                int  col = (int)n0 + wn * 32 + ni * 16 + lrow;
                float v = acc[mi][ni][r];
                long o = row * N + col;
                if (EPI == 0) outb[o] = (__bf16)v;
                if (EPI == 1) outb[o] = (__bf16)((v + bias[col]) * ps);
                if (EPI == 2) outb[o] = (__bf16)gelu_f(v + bias[col]);
                if (EPI == 3) outb[o] = (__bf16)gelu_f(v * sc[col] + sh[col]);
                if (EPI == 4) outf[o] += v + bias[col];
                if (EPI == 5) outf[o] += v;
            }
}

// ================= depthwise conv 3x3 + BN2 + GELU =================
__global__ void dwconv_kernel(const __bf16* __restrict__ Hin,
                              const float* __restrict__ W3,
                              const float* __restrict__ sc, const float* __restrict__ sh,
                              __bf16* __restrict__ Hout) {
    long idx = (long)blockIdx.x * 256 + threadIdx.x;
    int ce = (int)(idx & (CE - 1));
    long p = idx >> 10;
    int x = (int)(p & 63); long t = p >> 6;
    int y = (int)(t & 63); int b = (int)(t >> 6);
    float s = 0.f;
#pragma unroll
    for (int ky = 0; ky < 3; ++ky) {
        int yy = y + ky - 1; if ((unsigned)yy >= 64u) continue;
#pragma unroll
        for (int kx = 0; kx < 3; ++kx) {
            int xx = x + kx - 1; if ((unsigned)xx >= 64u) continue;
            s += (float)Hin[(((long)b * 64 + yy) * 64 + xx) * CE + ce] *
                 W3[(ky * 3 + kx) * CE + ce];
        }
    }
    Hout[idx] = (__bf16)gelu_f(s * sc[ce] + sh[ce]);
}

// ================= squeeze-excite =================
__global__ void sepool_kernel(const __bf16* __restrict__ H, float* __restrict__ P) {
    int b = blockIdx.y;
    int ce = blockIdx.x * 256 + threadIdx.x;
    const __bf16* p = H + (long)b * 4096 * CE + ce;
    float s = 0.f;
    for (int i = 0; i < 4096; ++i) s += (float)p[(long)i * CE];
    P[b * CE + ce] = s * (1.0f / 4096.0f);
}

__global__ void semlp_kernel(const float* __restrict__ P, const float* __restrict__ w1,
                             const float* __restrict__ b1, const float* __restrict__ w2,
                             const float* __restrict__ b2, float* __restrict__ G) {
    __shared__ float s1[16 * 64];
    for (int i = threadIdx.x; i < 16 * 64; i += 256) {
        int b = i >> 6, j = i & 63;
        float s = b1[j];
        for (int k = 0; k < CE; ++k) s += P[b * CE + k] * w1[k * 64 + j];
        s1[i] = gelu_f(s);
    }
    __syncthreads();
    for (int i = threadIdx.x; i < 16 * CE; i += 256) {
        int b = i >> 10, j = i & (CE - 1);
        float s = b2[j];
        for (int k = 0; k < 64; ++k) s += s1[b * 64 + k] * w2[k * CE + j];
        G[i] = 1.0f / (1.0f + __expf(-s));
    }
}

__global__ void segate_kernel(const __bf16* __restrict__ H, const float* __restrict__ G,
                              __bf16* __restrict__ O) {
    long idx = (long)blockIdx.x * 256 + threadIdx.x;
    int ce = (int)(idx & (CE - 1));
    int b = (int)(idx >> 22);   // / (4096*1024)
    O[idx] = (__bf16)((float)H[idx] * G[b * CE + ce]);
}

// ================= LayerNorm (wave32 shuffle reduce) =================
__global__ void ln_kernel(const float* __restrict__ X, const float* __restrict__ g,
                          const float* __restrict__ b, __bf16* __restrict__ Y) {
    long row = (long)blockIdx.x * 8 + (threadIdx.x >> 5);
    int lane = threadIdx.x & 31;
    const float* x = X + row * CCH;
    float v[8]; float s = 0.f;
#pragma unroll
    for (int j = 0; j < 8; ++j) { v[j] = x[j * 32 + lane]; s += v[j]; }
#pragma unroll
    for (int o = 16; o > 0; o >>= 1) s += __shfl_xor(s, o, 32);
    float mu = s * (1.0f / 256.0f);
    float q = 0.f;
#pragma unroll
    for (int j = 0; j < 8; ++j) { float d = v[j] - mu; q += d * d; }
#pragma unroll
    for (int o = 16; o > 0; o >>= 1) q += __shfl_xor(q, o, 32);
    float rs = rsqrtf(q * (1.0f / 256.0f) + 1e-5f);
#pragma unroll
    for (int j = 0; j < 8; ++j) {
        int c = j * 32 + lane;
        Y[row * CCH + c] = (__bf16)((v[j] - mu) * rs * g[c] + b[c]);
    }
}

// ================= window/grid attention (WMMA) =================
// one (window, head) per 64-thread block. GRIDMODE: 0 = window part, 1 = dilated grid.
template <int GRIDMODE>
__global__ __launch_bounds__(64) void attn_kernel(const __bf16* __restrict__ Q,
                                                  const __bf16* __restrict__ K,
                                                  const __bf16* __restrict__ V,
                                                  __bf16* __restrict__ O) {
    __shared__ __bf16 Qs[64 * 32], Ks[64 * 32], Vt[32 * 64];
    __shared__ float  Ss[64 * 64];
    __shared__ __bf16 Pb[64 * 64];
    __shared__ int tok[64];

    int head = blockIdx.x & 7;
    int win  = blockIdx.x >> 3;
    int t = threadIdx.x;
    int b = win >> 6, wi = win & 63, wr = wi >> 3, wc = wi & 7;
    int r, c;
    if (GRIDMODE) { r = (t >> 3) * 8 + wr; c = (t & 7) * 8 + wc; }
    else          { r = wr * 8 + (t >> 3); c = wc * 8 + (t & 7); }
    int g = (b * 64 + r) * 64 + c;          // global token (pixel) index
    tok[t] = g;
    const int co = head * HD;
#pragma unroll
    for (int d = 0; d < HD; ++d) {
        Qs[t * 32 + d] = Q[(long)g * CCH + co + d];
        Ks[t * 32 + d] = K[(long)g * CCH + co + d];
        Vt[d * 64 + t] = V[(long)g * CCH + co + d];  // V transposed tile
    }
    __syncthreads();

    const int wv = t >> 5, lane = t & 31;
    const int lrow = lane & 15, lhi = lane >> 4;

    // S = Q * K^T  (K-dim = 32 -> one WMMA per 16x16 tile)
    for (int qi = wv * 2; qi < wv * 2 + 2; ++qi) {
        v16bf a = ldfragA(&Qs[(qi * 16 + lrow) * 32 + lhi * 8]);
#pragma unroll
        for (int ki = 0; ki < 4; ++ki) {
            v16bf bb = ldfragB(&Ks[(ki * 16 + lrow) * 32 + lhi * 8]);
            v8f acc = {};
            acc = wmma_bf16(a, bb, acc);
#pragma unroll
            for (int r8 = 0; r8 < 8; ++r8)
                Ss[(qi * 16 + r8 + 8 * lhi) * 64 + ki * 16 + lrow] = acc[r8];
        }
    }
    __syncthreads();

    // softmax: one row per thread
    {
        float mx = -1e30f;
        for (int j = 0; j < 64; ++j) mx = fmaxf(mx, Ss[t * 64 + j]);
        float sum = 0.f;
        for (int j = 0; j < 64; ++j) { float e = __expf(Ss[t * 64 + j] - mx); sum += e; Ss[t * 64 + j] = e; }
        float inv = 1.0f / sum;
        for (int j = 0; j < 64; ++j) Pb[t * 64 + j] = (__bf16)(Ss[t * 64 + j] * inv);
    }
    __syncthreads();

    // O = P * V   (K-dim = 64 -> two WMMAs per 16x16 tile)
    for (int qi = wv * 2; qi < wv * 2 + 2; ++qi) {
#pragma unroll
        for (int oi = 0; oi < 2; ++oi) {
            v8f acc = {};
#pragma unroll
            for (int kc = 0; kc < 2; ++kc) {
                v16bf a  = ldfragA(&Pb[(qi * 16 + lrow) * 64 + kc * 32 + lhi * 8]);
                v16bf bb = ldfragB(&Vt[(oi * 16 + lrow) * 64 + kc * 32 + lhi * 8]);
                acc = wmma_bf16(a, bb, acc);
            }
#pragma unroll
            for (int r8 = 0; r8 < 8; ++r8) {
                int row = qi * 16 + r8 + 8 * lhi;
                O[(long)tok[row] * CCH + co + oi * 16 + lrow] = (__bf16)acc[r8];
            }
        }
    }
}

// ================= launch =================
extern "C" void kernel_launch(void* const* d_in, const int* in_sizes, int n_in,
                              void* d_out, int out_size, void* d_ws, size_t ws_size,
                              hipStream_t stream) {
    (void)in_sizes; (void)n_in; (void)out_size; (void)ws_size;
    const float* X = (const float*)d_in[0];
    float* OUT = (float*)d_out;                 // fp32 residual stream == output

    char* base = (char*)d_ws; size_t off = 0;
    auto alloc = [&](size_t bytes) -> void* {
        void* p = base + off; off += (bytes + 255) & ~(size_t)255; return p;
    };

    __bf16* xb   = (__bf16*)alloc((size_t)TOK * CCH * 2);
    __bf16* hln  = (__bf16*)alloc((size_t)TOK * CCH * 2);
    __bf16* qb   = (__bf16*)alloc((size_t)TOK * CCH * 2);
    __bf16* kb   = (__bf16*)alloc((size_t)TOK * CCH * 2);
    __bf16* vb   = (__bf16*)alloc((size_t)TOK * CCH * 2);
    __bf16* ab   = (__bf16*)alloc((size_t)TOK * CCH * 2);
    __bf16* wide1 = (__bf16*)alloc((size_t)TOK * CE * 2);   // expand / ffn hidden
    __bf16* wide2 = (__bf16*)alloc((size_t)TOK * CE * 2);   // dwconv out
    __bf16* wide3 = (__bf16*)alloc((size_t)TOK * CE * 2);   // SE-gated
    __bf16* expT = (__bf16*)alloc((size_t)CCH * CE * 2);
    __bf16* prjT = (__bf16*)alloc((size_t)CE * CCH * 2);
    float* sc1 = (float*)alloc(CE * 4); float* sh1 = (float*)alloc(CE * 4);
    float* sc2 = (float*)alloc(CE * 4); float* sh2 = (float*)alloc(CE * 4);
    float* pool = (float*)alloc(16 * CE * 4);
    float* gate = (float*)alloc(16 * CE * 4);
    __bf16* qT[2], *kT[2], *vT[2], *oT[2], *f1T[2], *f2T[2];
    for (int i = 0; i < 2; ++i) {
        qT[i] = (__bf16*)alloc((size_t)CCH * CCH * 2);
        kT[i] = (__bf16*)alloc((size_t)CCH * CCH * 2);
        vT[i] = (__bf16*)alloc((size_t)CCH * CCH * 2);
        oT[i] = (__bf16*)alloc((size_t)CCH * CCH * 2);
        f1T[i] = (__bf16*)alloc((size_t)CCH * CE * 2);
        f2T[i] = (__bf16*)alloc((size_t)CE * CCH * 2);
    }

    auto tc = [&](const void* in, __bf16* outp, int K, int N) {
        tcast_kernel<<<((long)K * N + 255) / 256, 256, 0, stream>>>((const float*)in, outp, K, N);
    };

    // ---- weight prep ----
    tc(d_in[5], expT, CCH, CE);
    tc(d_in[19], prjT, CE, CCH);
    bnfold_kernel<<<4, 256, 0, stream>>>((const float*)d_in[6], (const float*)d_in[7],
                                         (const float*)d_in[8], (const float*)d_in[9], sc1, sh1, CE);
    bnfold_kernel<<<4, 256, 0, stream>>>((const float*)d_in[11], (const float*)d_in[12],
                                         (const float*)d_in[13], (const float*)d_in[14], sc2, sh2, CE);

    // ---- MBConv ----
    bn0cast_kernel<<<(long)TOK * CCH / 256, 256, 0, stream>>>(
        X, (const float*)d_in[1], (const float*)d_in[2], (const float*)d_in[3],
        (const float*)d_in[4], xb, OUT);
    gemm_bf16<3><<<dim3(CE / 64, TOK / 128), 256, 0, stream>>>(
        xb, expT, nullptr, sc1, sh1, wide1, nullptr, CCH, CE, 1.0f);
    dwconv_kernel<<<(long)TOK * CE / 256, 256, 0, stream>>>(
        wide1, (const float*)d_in[10], sc2, sh2, wide2);
    sepool_kernel<<<dim3(CE / 256, 16), 256, 0, stream>>>(wide2, pool);
    semlp_kernel<<<1, 256, 0, stream>>>(pool, (const float*)d_in[15], (const float*)d_in[16],
                                        (const float*)d_in[17], (const float*)d_in[18], gate);
    segate_kernel<<<(long)TOK * CE / 256, 256, 0, stream>>>(wide2, gate, wide3);
    gemm_bf16<5><<<dim3(CCH / 64, TOK / 128), 256, 0, stream>>>(
        wide3, prjT, nullptr, nullptr, nullptr, nullptr, OUT, CE, CCH, 1.0f);

    // ---- attention stage ----
    auto attn_stage = [&](int pb, int gm, int slot) {
        tc(d_in[pb + 2], qT[slot], CCH, CCH);
        tc(d_in[pb + 4], kT[slot], CCH, CCH);
        tc(d_in[pb + 6], vT[slot], CCH, CCH);
        tc(d_in[pb + 8], oT[slot], CCH, CCH);
        ln_kernel<<<TOK / 8, 256, 0, stream>>>(OUT, (const float*)d_in[pb],
                                               (const float*)d_in[pb + 1], hln);
        gemm_bf16<1><<<dim3(CCH / 64, TOK / 128), 256, 0, stream>>>(
            hln, qT[slot], (const float*)d_in[pb + 3], nullptr, nullptr, qb, nullptr,
            CCH, CCH, 0.17677669529663689f);   // HD^-0.5
        gemm_bf16<1><<<dim3(CCH / 64, TOK / 128), 256, 0, stream>>>(
            hln, kT[slot], (const float*)d_in[pb + 5], nullptr, nullptr, kb, nullptr, CCH, CCH, 1.0f);
        gemm_bf16<1><<<dim3(CCH / 64, TOK / 128), 256, 0, stream>>>(
            hln, vT[slot], (const float*)d_in[pb + 7], nullptr, nullptr, vb, nullptr, CCH, CCH, 1.0f);
        if (gm) attn_kernel<1><<<1024 * NHD, 64, 0, stream>>>(qb, kb, vb, ab);
        else    attn_kernel<0><<<1024 * NHD, 64, 0, stream>>>(qb, kb, vb, ab);
        gemm_bf16<4><<<dim3(CCH / 64, TOK / 128), 256, 0, stream>>>(
            ab, oT[slot], (const float*)d_in[pb + 9], nullptr, nullptr, nullptr, OUT, CCH, CCH, 1.0f);
    };

    auto ffn_stage = [&](int pb, int slot) {
        tc(d_in[pb + 2], f1T[slot], CCH, CE);
        tc(d_in[pb + 4], f2T[slot], CE, CCH);
        ln_kernel<<<TOK / 8, 256, 0, stream>>>(OUT, (const float*)d_in[pb],
                                               (const float*)d_in[pb + 1], hln);
        gemm_bf16<2><<<dim3(CE / 64, TOK / 128), 256, 0, stream>>>(
            hln, f1T[slot], (const float*)d_in[pb + 3], nullptr, nullptr, wide1, nullptr, CCH, CE, 1.0f);
        gemm_bf16<4><<<dim3(CCH / 64, TOK / 128), 256, 0, stream>>>(
            wide1, f2T[slot], (const float*)d_in[pb + 5], nullptr, nullptr, nullptr, OUT, CE, CCH, 1.0f);
    };

    attn_stage(20, 0, 0);   // block (window) attention
    ffn_stage(30, 0);
    attn_stage(36, 1, 1);   // grid (dilated) attention
    ffn_stage(46, 1);
}
